// CustomFlashAttention_39324720562719
// MI455X (gfx1250) — matrix operand order repaired and checked by reading, not compiled, not run on hardware
//
#include <hip/hip_runtime.h>
#include <hip/hip_bf16.h>

typedef __attribute__((ext_vector_type(16))) __bf16 v16bf;
typedef __attribute__((ext_vector_type(8)))  float  v8f;

union Frag16 {
  v16bf v;
  uint4 q[2];
};

__device__ __forceinline__ v8f v8f_zero() {
  v8f r;
#pragma unroll
  for (int i = 0; i < 8; ++i) r[i] = 0.0f;
  return r;
}

__device__ __forceinline__ v8f wmma_bf16(v16bf a, v16bf b, v8f c) {
  // (neg_a, A, neg_b, B, c_mod, C, reuse_a, reuse_b)
  return __builtin_amdgcn_wmma_f32_16x16x32_bf16(false, a, false, b, (short)0, c,
                                                 false, false);
}

// ---------------------------------------------------------------------------
// f32 -> bf16 conversion (vectorized)
// ---------------------------------------------------------------------------
__global__ void cvt_f32_bf16(const float* __restrict__ in,
                             __bf16* __restrict__ out, int n4) {
  int i = blockIdx.x * blockDim.x + threadIdx.x;
  if (i >= n4) return;
  float4 f = ((const float4*)in)[i];
  union { __bf16 b[4]; uint2 u; } o;
  o.b[0] = (__bf16)f.x; o.b[1] = (__bf16)f.y;
  o.b[2] = (__bf16)f.z; o.b[3] = (__bf16)f.w;
  ((uint2*)out)[i] = o.u;
}

// ---------------------------------------------------------------------------
// C = A (MxK) @ W^T (W is NxK row-major)  -- bf16 WMMA, f32 accumulate.
// Block: 256 threads / 8 waves, 128x128 output tile, BK=32.
// Wave grid 4(m) x 2(n); each wave: 2x4 tiles of 16x16.
// ---------------------------------------------------------------------------
constexpr int BM = 128, BN = 128, BK = 32, LDT = BK + 8;

template <bool STORE_BF16>
__global__ __launch_bounds__(256) void gemm_xwT(const __bf16* __restrict__ A,
                                                const __bf16* __restrict__ W,
                                                void* __restrict__ C, int M,
                                                int N, int K) {
  __shared__ __bf16 sA[BM * LDT];
  __shared__ __bf16 sB[BN * LDT];

  const int tid = threadIdx.x;
  const int lane = tid & 31;
  const int wid = tid >> 5;
  const int wm = wid & 3;
  const int wn = wid >> 2;
  const int m0 = blockIdx.x * BM;
  const int n0 = blockIdx.y * BN;
  const int hi = lane >> 4;   // lane half selects contiguous K half (16*hi .. 16*hi+15)
  const int l15 = lane & 15;

  v8f acc[2][4];
#pragma unroll
  for (int i = 0; i < 2; ++i)
#pragma unroll
    for (int j = 0; j < 4; ++j) acc[i][j] = v8f_zero();

  // staging: each thread copies 16 bf16 (32B) of A and of W per K step
  const int srow = tid >> 1;
  const int soff = (tid & 1) * 16;
  const __bf16* gA = A + (size_t)(m0 + srow) * K + soff;
  const __bf16* gW = W + (size_t)(n0 + srow) * K + soff;
  __bf16* lA = sA + srow * LDT + soff;
  __bf16* lB = sB + srow * LDT + soff;

  for (int k0 = 0; k0 < K; k0 += BK) {
    __syncthreads();
    uint4 a0 = ((const uint4*)(gA + k0))[0];
    uint4 a1 = ((const uint4*)(gA + k0))[1];
    uint4 b0 = ((const uint4*)(gW + k0))[0];
    uint4 b1 = ((const uint4*)(gW + k0))[1];
    ((uint4*)lA)[0] = a0; ((uint4*)lA)[1] = a1;
    ((uint4*)lB)[0] = b0; ((uint4*)lB)[1] = b1;
    if (k0 + BK < K) {  // global_prefetch_b8 for next K tile
      __builtin_prefetch(gA + k0 + BK, 0, 0);
      __builtin_prefetch(gW + k0 + BK, 0, 0);
    }
    __syncthreads();

    // A fragments: lane 0-15 row m, K 0-15; lane 16-31 K 16-31 (contig)
    Frag16 af[2];
#pragma unroll
    for (int i = 0; i < 2; ++i) {
      const __bf16* p = sA + (wm * 32 + i * 16 + l15) * LDT + hi * 16;
      af[i].q[0] = *(const uint4*)p;
      af[i].q[1] = *(const uint4*)(p + 8);
    }
#pragma unroll
    for (int j = 0; j < 4; ++j) {
      // B fragment: lane half 0 holds K 0-15, half 1 holds K 16-31 (contig)
      const __bf16* p = sB + (wn * 64 + j * 16 + l15) * LDT + hi * 16;
      Frag16 bf;
      bf.q[0] = ((const uint4*)p)[0];
      bf.q[1] = ((const uint4*)p)[1];
#pragma unroll
      for (int i = 0; i < 2; ++i) acc[i][j] = wmma_bf16(af[i].v, bf.v, acc[i][j]);
    }
  }

  // C layout: VGPR r -> row r (lanes 0-15) / row r+8 (lanes 16-31), col = l15
#pragma unroll
  for (int i = 0; i < 2; ++i)
#pragma unroll
    for (int j = 0; j < 4; ++j) {
      const int col = n0 + wn * 64 + j * 16 + l15;
#pragma unroll
      for (int r = 0; r < 8; ++r) {
        const int row = m0 + wm * 32 + i * 16 + r + hi * 8;
        const float v = acc[i][j][r];
        if (STORE_BF16)
          ((__bf16*)C)[(size_t)row * N + col] = (__bf16)v;
        else
          ((float*)C)[(size_t)row * N + col] = v;
      }
    }
}

// ---------------------------------------------------------------------------
// Causal flash attention. Q/K/V/O: [B, S, H*HD] bf16.
// Block = 128 threads (4 waves) = 64 query rows of one (b, h).
// Each wave owns 16 query rows; 32 keys per iteration.
// ---------------------------------------------------------------------------
constexpr int S_DIM = 2048, D_DIM = 2048, HD = 128;

__global__ __launch_bounds__(128) void flash_attn(
    const __bf16* __restrict__ Q, const __bf16* __restrict__ Kb,
    const __bf16* __restrict__ V, __bf16* __restrict__ O,
    const int* __restrict__ causal_p) {
  __shared__ __bf16 kt[32][HD + 8];      // K rows (direct copy)
  __shared__ __bf16 vt[HD][32 + 8];      // V transposed (d-major)
  __shared__ __bf16 pt[4][16][32 + 8];   // per-wave P re-layout buffer

  const int b = blockIdx.z;
  const int h = blockIdx.y;
  const int qb = blockIdx.x;
  const int tid = threadIdx.x;
  const int lane = tid & 31;
  const int w = tid >> 5;
  const int hi = lane >> 4;
  const int l15 = lane & 15;
  const int qm0 = qb * 64 + w * 16;
  const int causal = *causal_p;

  const size_t base = ((size_t)b * S_DIM) * D_DIM + (size_t)h * HD;

  // Preload Q as 4 A-fragments (d split into 4 chunks of 32)
  Frag16 qf[4];
#pragma unroll
  for (int c = 0; c < 4; ++c) {
    const __bf16* p = Q + base + (size_t)(qm0 + l15) * D_DIM + c * 32 + hi * 16;
    qf[c].q[0] = *(const uint4*)p;
    qf[c].q[1] = *(const uint4*)(p + 8);
  }

  v8f oacc[8];
#pragma unroll
  for (int j = 0; j < 8; ++j) oacc[j] = v8f_zero();
  float mrow[8], lrow[8];
#pragma unroll
  for (int r = 0; r < 8; ++r) {
    mrow[r] = -__builtin_inff();
    lrow[r] = 0.0f;
  }

  const float sc = 0.0883883476483184f;  // 1/sqrt(128)
  const int kend = causal ? (qb * 64 + 64) : S_DIM;

  for (int kn0 = 0; kn0 < kend; kn0 += 32) {
    __syncthreads();
    {  // stage 32x128 K tile: thread copies 64B of one row
      const int r = tid >> 2;
      const int cb = (tid & 3) * 32;
      const uint4* g = (const uint4*)(Kb + base + (size_t)(kn0 + r) * D_DIM + cb);
      uint4* s = (uint4*)&kt[r][cb];
      s[0] = g[0]; s[1] = g[1]; s[2] = g[2]; s[3] = g[3];
    }
    {  // stage V transposed: coalesced read along d, scatter into vt[d][key]
      const int kl = tid & 31;
      const int db = (tid >> 5) * 32;
      const __bf16* g = V + base + (size_t)(kn0 + kl) * D_DIM + db;
#pragma unroll
      for (int d = 0; d < 32; ++d) vt[db + d][kl] = g[d];
    }
    __syncthreads();

    if (causal && (kn0 > qm0 + 15)) continue;  // fully-masked tile for wave

    // S = Q K^T for keys [kn0, kn0+32): two 16x16 tiles, 4 K-chunks each
    v8f s0 = v8f_zero(), s1 = v8f_zero();
#pragma unroll
    for (int c = 0; c < 4; ++c) {
      Frag16 b0, b1;
      const __bf16* p0 = &kt[l15][c * 32 + hi * 16];
      const __bf16* p1 = &kt[16 + l15][c * 32 + hi * 16];
      b0.q[0] = ((const uint4*)p0)[0]; b0.q[1] = ((const uint4*)p0)[1];
      b1.q[0] = ((const uint4*)p1)[0]; b1.q[1] = ((const uint4*)p1)[1];
      s0 = wmma_bf16(qf[c].v, b0.v, s0);
      s1 = wmma_bf16(qf[c].v, b1.v, s1);
    }

    // Online softmax. Row r lives in VGPR r across one 16-lane half.
    float alpha[8];
#pragma unroll
    for (int r = 0; r < 8; ++r) {
      const int qrow = qm0 + r + hi * 8;
      float x0 = s0[r] * sc;
      float x1 = s1[r] * sc;
      if (causal) {
        if (kn0 + l15 > qrow)      x0 = -__builtin_inff();
        if (kn0 + 16 + l15 > qrow) x1 = -__builtin_inff();
      }
      float t = fmaxf(x0, x1);
      t = fmaxf(t, __shfl_xor(t, 1));
      t = fmaxf(t, __shfl_xor(t, 2));
      t = fmaxf(t, __shfl_xor(t, 4));
      t = fmaxf(t, __shfl_xor(t, 8));  // stays within 16-lane half (wave32)
      const float mn = fmaxf(mrow[r], t);
      alpha[r] = __expf(mrow[r] - mn);
      mrow[r] = mn;
      const float e0 = __expf(x0 - mn);
      const float e1 = __expf(x1 - mn);
      float su = e0 + e1;
      su += __shfl_xor(su, 1);
      su += __shfl_xor(su, 2);
      su += __shfl_xor(su, 4);
      su += __shfl_xor(su, 8);
      lrow[r] = lrow[r] * alpha[r] + su;
      // park P in LDS (C layout -> memory); same-wave DS ops are in-order
      pt[w][r + hi * 8][l15] = (__bf16)e0;
      pt[w][r + hi * 8][16 + l15] = (__bf16)e1;
    }
#pragma unroll
    for (int j = 0; j < 8; ++j)
#pragma unroll
      for (int r = 0; r < 8; ++r) oacc[j][r] *= alpha[r];

    // Reload P as an A-fragment (16x32 over keys)
    Frag16 pf;
    const __bf16* pp = &pt[w][l15][hi * 16];
    pf.q[0] = *(const uint4*)pp;
    pf.q[1] = *(const uint4*)(pp + 8);

    // O += P V : 8 d-chunks of 16 columns
#pragma unroll
    for (int j = 0; j < 8; ++j) {
      Frag16 vf;
      const __bf16* vp = &vt[j * 16 + l15][hi * 16];
      vf.q[0] = ((const uint4*)vp)[0];
      vf.q[1] = ((const uint4*)vp)[1];
      oacc[j] = wmma_bf16(pf.v, vf.v, oacc[j]);
    }
  }

#pragma unroll
  for (int r = 0; r < 8; ++r) lrow[r] = 1.0f / lrow[r];
#pragma unroll
  for (int j = 0; j < 8; ++j) {
    const int col = h * HD + j * 16 + l15;
#pragma unroll
    for (int r = 0; r < 8; ++r) {
      const int row = qm0 + r + hi * 8;
      O[((size_t)b * S_DIM + row) * D_DIM + col] = (__bf16)(oacc[j][r] * lrow[r]);
    }
  }
}

// ---------------------------------------------------------------------------
extern "C" void kernel_launch(void* const* d_in, const int* in_sizes, int n_in,
                              void* d_out, int out_size, void* d_ws,
                              size_t ws_size, hipStream_t stream) {
  (void)in_sizes; (void)n_in; (void)out_size; (void)ws_size;
  constexpr int B = 2, S = 2048, D = 2048;
  const float* x = (const float*)d_in[0];
  const float* wq = (const float*)d_in[1];
  const float* wk = (const float*)d_in[2];
  const float* wv = (const float*)d_in[3];
  const float* wo = (const float*)d_in[4];
  const int* causal = (const int*)d_in[5];
  float* out = (float*)d_out;

  const size_t nx = (size_t)B * S * D;  // 8,388,608
  const size_t nw = (size_t)D * D;      // 4,194,304

  char* p = (char*)d_ws;
  auto carve = [&](size_t elems) {
    __bf16* r = (__bf16*)p;
    p += ((elems * sizeof(__bf16)) + 255) & ~(size_t)255;
    return r;
  };
  __bf16* xb = carve(nx);
  __bf16* wqb = carve(nw);
  __bf16* wkb = carve(nw);
  __bf16* wvb = carve(nw);
  __bf16* wob = carve(nw);
  __bf16* qb = carve(nx);
  __bf16* kb = carve(nx);
  __bf16* vb = carve(nx);
  __bf16* ob = carve(nx);

  const int T = 256;
  cvt_f32_bf16<<<dim3((unsigned)((nx / 4 + T - 1) / T)), T, 0, stream>>>(x, xb, (int)(nx / 4));
  cvt_f32_bf16<<<dim3((unsigned)((nw / 4 + T - 1) / T)), T, 0, stream>>>(wq, wqb, (int)(nw / 4));
  cvt_f32_bf16<<<dim3((unsigned)((nw / 4 + T - 1) / T)), T, 0, stream>>>(wk, wkb, (int)(nw / 4));
  cvt_f32_bf16<<<dim3((unsigned)((nw / 4 + T - 1) / T)), T, 0, stream>>>(wv, wvb, (int)(nw / 4));
  cvt_f32_bf16<<<dim3((unsigned)((nw / 4 + T - 1) / T)), T, 0, stream>>>(wo, wob, (int)(nw / 4));

  dim3 gg(B * S / BM, D / BN);  // 32 x 16
  gemm_xwT<true><<<gg, 256, 0, stream>>>(xb, wqb, qb, B * S, D, D);
  gemm_xwT<true><<<gg, 256, 0, stream>>>(xb, wkb, kb, B * S, D, D);
  gemm_xwT<true><<<gg, 256, 0, stream>>>(xb, wvb, vb, B * S, D, D);

  flash_attn<<<dim3(S / 64, 16, B), 128, 0, stream>>>(qb, kb, vb, ob, causal);

  gemm_xwT<false><<<gg, 256, 0, stream>>>(ob, wob, out, B * S, D, D);
}
